// ResidualLFQ_62431644615312
// MI455X (gfx1250) — matrix operand order repaired and checked by reading, not compiled, not run on hardware
//
#include <hip/hip_runtime.h>

typedef __attribute__((ext_vector_type(16))) _Float16 v16h;
typedef __attribute__((ext_vector_type(8)))  float    v8f;
typedef __attribute__((ext_vector_type(4)))  float    v4f;

#define BN        65536            // 16 * 4096 samples
#define DIM       512
#define CBD       10               // codebook dim
#define NQ        8                // quantizers
#define OUT0      ((size_t)0)
#define IDX0      ((size_t)BN * DIM)            // 33554432
#define LOSS0     (IDX0 + (size_t)BN * NQ)      // 34078720

// d_ws layout (bytes):
//   [0      .. 32768)  gAvg  : float[8][1024] avg_prob accumulators
//   [32768  .. 32832)  sums  : float[16]  (H partial [0..7], commit partial [8..15])
//   [32832  .. 49216)  winB  : _Float16[16*32*16]  B-operand tiles for w_in (K=512, N=16 padded)
//   [49216  .. 81984)  woutB : _Float16[32*32*16]  B-operand tiles for w_out (K=32 padded, N=512)

// ---------------------------------------------------------------------------
// Kernel 0: zero accumulators + pre-swizzle w_in / w_out into a contiguous-k
// wave32 WMMA B-operand layout (per (tile,lane): 16 contiguous halves).
// B k order used here (A matches): lane<16 -> N=lane,    K = kbase + e
//                            lane>=16-> N=lane-16, K = kbase + 16 + e
// ---------------------------------------------------------------------------
__global__ void rlfq_setup(const float* __restrict__ w_in,
                           const float* __restrict__ w_out,
                           float* __restrict__ gAvg, float* __restrict__ sums,
                           _Float16* __restrict__ winB, _Float16* __restrict__ woutB) {
  int tid = blockIdx.x * blockDim.x + threadIdx.x;
  if (tid < NQ * 1024) gAvg[tid] = 0.0f;
  if (tid < 16) sums[tid] = 0.0f;
  if (tid < 16 * 32) {                       // winB: 16 k-steps x 32 lanes
    int kstep = tid >> 5, lane = tid & 31;
    int n = lane & 15;
    int kbase = kstep * 32 + ((lane < 16) ? 0 : 16);
    for (int e = 0; e < 16; ++e) {
      int K = kbase + e;
      float v = (n < CBD) ? w_in[K * CBD + n] : 0.0f;
      winB[(size_t)tid * 16 + e] = (_Float16)v;
    }
  }
  if (tid < 32 * 32) {                       // woutB: 32 n-tiles x 32 lanes
    int nt = tid >> 5, lane = tid & 31;
    int n = nt * 16 + (lane & 15);
    int kbase = (lane < 16) ? 0 : 16;
    for (int e = 0; e < 16; ++e) {
      int K = kbase + e;
      float v = (K < CBD) ? w_out[K * DIM + n] : 0.0f;
      woutB[(size_t)tid * 16 + e] = (_Float16)v;
    }
  }
}

// ---------------------------------------------------------------------------
// Kernel 1: fused projection (WMMA) -> 8x LFQ (closed-form entropy) ->
//           output projection (WMMA). 256 threads = 8 waves, 256 samples/block.
// Each wave owns samples [32w, 32w+32) == its own threads' rows in LDS.
// ---------------------------------------------------------------------------
__global__ __launch_bounds__(256) void rlfq_main(
    const float* __restrict__ x, const float* __restrict__ b_in,
    const float* __restrict__ b_out,
    const _Float16* __restrict__ winB, const _Float16* __restrict__ woutB,
    float* __restrict__ gAvg, float* __restrict__ sums,
    float* __restrict__ out) {
  __shared__ float sBuf[256][12];            // h -> pPlus -> qsum staging

  const int t     = threadIdx.x;
  const int lane  = t & 31;
  const int w     = t >> 5;
  const int n16   = lane & 15;
  const int lhalf = (lane < 16) ? 0 : 1;
  const int blockStart = blockIdx.x * 256;

  // ---------------- Phase 1: h = x @ w_in + b_in  (f16 WMMA, f32 accum) ----
  for (int tile = 2 * w; tile <= 2 * w + 1; ++tile) {
    const int row = blockStart + tile * 16 + n16;     // A-layout: M = lane&15
    const float* xr = x + (size_t)row * DIM + lhalf * 16;
    v8f c = {};
    for (int k0 = 0; k0 < 16; ++k0) {
      const float* p0 = xr + k0 * 32;
      v4f f0 = *(const v4f*)(p0);
      v4f f1 = *(const v4f*)(p0 + 4);
      v4f f2 = *(const v4f*)(p0 + 8);
      v4f f3 = *(const v4f*)(p0 + 12);
      v16h a;
#pragma unroll
      for (int e = 0; e < 4; ++e) {
        a[e]      = (_Float16)f0[e];
        a[e + 4]  = (_Float16)f1[e];
        a[e + 8]  = (_Float16)f2[e];
        a[e + 12] = (_Float16)f3[e];
      }
      v16h b = *(const v16h*)(winB + ((size_t)(k0 * 32 + lane)) * 16);
      c = __builtin_amdgcn_wmma_f32_16x16x32_f16(false, a, false, b,
                                                 (short)0, c, false, false);
    }
    if (n16 < CBD) {                         // D: VGPR r -> M = r (+8 hi half)
      float bi = b_in[n16];
#pragma unroll
      for (int r = 0; r < 8; ++r)
        sBuf[tile * 16 + r + lhalf * 8][n16] = c[r] + bi;
    }
  }
  __syncthreads();

  // ---------------- Phase 2: residual LFQ, one thread per sample -----------
  float r_[CBD], qsum[CBD];
#pragma unroll
  for (int d = 0; d < CBD; ++d) { r_[d] = sBuf[t][d]; qsum[d] = 0.0f; }
  __syncthreads();

  float scale = 1.0f;
  for (int ind = 0; ind < NQ; ++ind) {
    const float a2 = 200.0f * scale;         // z_c = a2 * (r . s_c)
    float Hloc = 0.0f, Cloc = 0.0f;
    int idx = 0;
#pragma unroll
    for (int d = 0; d < CBD; ++d) {
      bool pos = r_[d] > 0.0f;
      float q = pos ? scale : -scale;
      idx |= pos ? (1 << (9 - d)) : 0;
      float u  = a2 * r_[d];
      float au = fabsf(u);
      float e  = __expf(-2.0f * au);
      float inv1pe = 1.0f / (1.0f + e);
      // per-sample entropy: log(2cosh u) - u*tanh(u), stable form
      Hloc += au + log1pf(e) - au * ((1.0f - e) * inv1pe);
      float diff = r_[d] - q;
      Cloc += diff * diff;                   // commitment
      sBuf[t][d] = pos ? inv1pe : (e * inv1pe);   // pPlus = sigmoid(2u)
      r_[d] -= q;
      qsum[d] += q;
    }
    out[IDX0 + (size_t)(blockStart + t) * NQ + ind] = (float)idx;
    // wave-reduce H / commit, one atomic per wave
#pragma unroll
    for (int m = 16; m > 0; m >>= 1) {
      Hloc += __shfl_xor(Hloc, m);
      Cloc += __shfl_xor(Cloc, m);
    }
    if (lane == 0) {
      atomicAdd(&sums[ind], Hloc);
      atomicAdd(&sums[8 + ind], Cloc);
    }
    __syncthreads();

    // avg_prob accumulation: p_c factorizes over dims. Lane owns codes
    // c = lane*32 + j2 (lane bits -> dims 0..4, j2 bits -> dims 5..9).
    float acc[32];
#pragma unroll
    for (int j = 0; j < 32; ++j) acc[j] = 0.0f;
    for (int j = 0; j < 32; ++j) {           // 32 samples owned by this wave
      int sl = 32 * w + j;
      float pp[CBD];
#pragma unroll
      for (int d = 0; d < CBD; ++d) pp[d] = sBuf[sl][d];   // LDS broadcast
      float ph = 1.0f;
#pragma unroll
      for (int d = 0; d < 5; ++d) {
        int bit = (lane >> (4 - d)) & 1;
        ph *= bit ? pp[d] : (1.0f - pp[d]);
      }
      float plo[32];                         // subset-product DP over dims 5..9
      plo[0] = 1.0f;
      int sz = 1;
#pragma unroll
      for (int d = 9; d >= 5; --d) {
        float ppd = pp[d], pmd = 1.0f - ppd;
#pragma unroll
        for (int jj = 0; jj < 16; ++jj) {
          if (jj < sz) {
            plo[jj + sz] = plo[jj] * ppd;
            plo[jj]      = plo[jj] * pmd;
          }
        }
        sz <<= 1;
      }
#pragma unroll
      for (int j2 = 0; j2 < 32; ++j2) acc[j2] = fmaf(ph, plo[j2], acc[j2]);
    }
    float* g = gAvg + ind * 1024 + lane * 32;
#pragma unroll
    for (int j2 = 0; j2 < 32; ++j2) atomicAdd(&g[j2], acc[j2]);
    __syncthreads();
    scale *= 0.5f;
  }

#pragma unroll
  for (int d = 0; d < CBD; ++d) sBuf[t][d] = qsum[d];
  __syncthreads();

  // ---------------- Phase 3: out = qsum @ w_out + b_out (f16 WMMA) ---------
  for (int tile = 2 * w; tile <= 2 * w + 1; ++tile) {
    const int m = tile * 16 + n16;           // A row (uniform across halves)
    v16h a;
#pragma unroll
    for (int e = 0; e < 16; ++e) {
      float v = (lane < 16) ? ((e < 10) ? sBuf[m][e] : 0.0f)
                            : 0.0f;
      a[e] = (_Float16)v;
    }
    const int rowbase = blockStart + tile * 16 + lhalf * 8;
    for (int nt = 0; nt < 32; ++nt) {
      v16h b = *(const v16h*)(woutB + ((size_t)(nt * 32 + lane)) * 16);
      v8f c = {};
      c = __builtin_amdgcn_wmma_f32_16x16x32_f16(false, a, false, b,
                                                 (short)0, c, false, false);
      int n = nt * 16 + n16;
      float bo = b_out[n];
#pragma unroll
      for (int r = 0; r < 8; ++r)
        out[OUT0 + (size_t)(rowbase + r) * DIM + n] = c[r] + bo;
    }
  }
}

// ---------------------------------------------------------------------------
// Kernel 2: codebook entropy + final 8 losses
// ---------------------------------------------------------------------------
__global__ void rlfq_finalize(const float* __restrict__ gAvg,
                              const float* __restrict__ sums,
                              float* __restrict__ out) {
  __shared__ float sE[NQ];
  int t = threadIdx.x;                       // 256 threads
  if (t < NQ) sE[t] = 0.0f;
  __syncthreads();
  for (int ind = 0; ind < NQ; ++ind) {
    float local = 0.0f;
    for (int c = t; c < 1024; c += 256) {
      float p = gAvg[ind * 1024 + c] * (1.0f / 65536.0f);
      local -= p * __logf(fmaxf(p, 1e-20f));
    }
    atomicAdd(&sE[ind], local);
  }
  __syncthreads();
  if (t < NQ) {
    float Hmean = sums[t] * (1.0f / 65536.0f);
    float Cmean = sums[8 + t] * (1.0f / (65536.0f * 10.0f));
    out[LOSS0 + t] = 0.1f * (Hmean - sE[t]) + 0.25f * Cmean;
  }
}

// ---------------------------------------------------------------------------
extern "C" void kernel_launch(void* const* d_in, const int* in_sizes, int n_in,
                              void* d_out, int out_size, void* d_ws, size_t ws_size,
                              hipStream_t stream) {
  const float* x     = (const float*)d_in[0];
  const float* w_in  = (const float*)d_in[1];
  const float* b_in  = (const float*)d_in[2];
  const float* w_out = (const float*)d_in[3];
  const float* b_out = (const float*)d_in[4];
  float* out = (float*)d_out;

  float*    gAvg  = (float*)d_ws;
  float*    sums  = gAvg + NQ * 1024;
  _Float16* winB  = (_Float16*)((char*)d_ws + 32832);
  _Float16* woutB = (_Float16*)((char*)d_ws + 49216);

  rlfq_setup<<<32, 256, 0, stream>>>(w_in, w_out, gAvg, sums, winB, woutB);
  rlfq_main<<<BN / 256, 256, 0, stream>>>(x, b_in, b_out, winB, woutB,
                                          gAvg, sums, out);
  rlfq_finalize<<<1, 256, 0, stream>>>(gAvg, sums, out);
}
